// LightGlueTransformerLayer_65515431133859
// MI455X (gfx1250) — matrix operand order repaired and checked by reading, not compiled, not run on hardware
//
#include <hip/hip_runtime.h>
#include <math.h>

// ---------------------------------------------------------------------------
// Types for CDNA5 WMMA (gfx1250, wave32)
// ---------------------------------------------------------------------------
typedef __attribute__((ext_vector_type(16))) __bf16 v16bf;
typedef __attribute__((ext_vector_type(2)))  __bf16 bfx2;
typedef __attribute__((ext_vector_type(8)))  float  v8f;

#define BBATCH 4
#define NSEQ   2048
#define DMODEL 256
#define NHEAD  4
#define DHEAD  64
#define MROWS  (BBATCH * NSEQ)   // 8192

// ===========================================================================
// GEMM: C[M,N] = A[M,K] @ W[K,N] + bias[N] (+ resid[M,N])
// Tile 128x128x32, 256 threads = 8 waves (4x2), each wave 32x64 output.
// fp32 -> bf16 staged in LDS (B transposed); register double-buffered K loop;
// v_wmma_f32_16x16x32_bf16 with f32 accumulation.
// Requires M%128==0, N%128==0, K%32==0 (true for all shapes here).
// ===========================================================================
__global__ __launch_bounds__(256) void gemm_bias_kernel(
    const float* __restrict__ A, const float* __restrict__ W,
    const float* __restrict__ bias, const float* __restrict__ resid,
    float* __restrict__ C, int Kdim, int Ndim)
{
    __shared__ __bf16 As[128][40];   // row-major [r][k], padded
    __shared__ __bf16 Bs[128][40];   // transposed [n][k], padded

    const int tid  = threadIdx.x;
    const int wave = tid >> 5;
    const int lane = tid & 31;
    const int wm   = wave >> 1;          // 0..3
    const int wn   = wave & 1;           // 0..1
    const int bm   = blockIdx.y * 128;
    const int bn   = blockIdx.x * 128;
    const int half = lane >> 4;
    const int l16  = lane & 15;

    v8f acc[2][4];
#pragma unroll
    for (int mt = 0; mt < 2; ++mt)
#pragma unroll
        for (int nt = 0; nt < 4; ++nt)
            acc[mt][nt] = (v8f){0.f,0.f,0.f,0.f,0.f,0.f,0.f,0.f};

    // Staging registers (per-thread share of a 128x32 A tile / 32x128 B tile)
    float2 areg[8];
    float  blo[8], bhi[8];

    // --- load tile kk=0 ---
#pragma unroll
    for (int i = 0; i < 8; ++i) {
        int id = i * 256 + tid;
        int r  = id >> 4, c = (id & 15) * 2;          // A: 2 consecutive k
        areg[i] = *(const float2*)(A + (unsigned)(bm + r) * Kdim + c);
        int k2 = id >> 7, n = id & 127;               // B: k-pair at column n
        blo[i] = W[(unsigned)(2 * k2)     * Ndim + bn + n];
        bhi[i] = W[(unsigned)(2 * k2 + 1) * Ndim + bn + n];
    }

    for (int kk = 0; kk < Kdim; kk += 32) {
        __syncthreads();
        // Commit staged registers to LDS as packed bf16 pairs
#pragma unroll
        for (int i = 0; i < 8; ++i) {
            int id = i * 256 + tid;
            int r  = id >> 4, c = (id & 15) * 2;
            bfx2 pa; pa[0] = (__bf16)areg[i].x; pa[1] = (__bf16)areg[i].y;
            *(bfx2*)&As[r][c] = pa;
            int k2 = id >> 7, n = id & 127;
            bfx2 pb; pb[0] = (__bf16)blo[i]; pb[1] = (__bf16)bhi[i];
            *(bfx2*)&Bs[n][2 * k2] = pb;
        }
        __syncthreads();

        // Issue next tile's global loads BEFORE compute (latency overlap)
        if (kk + 32 < Kdim) {
            const int kn = kk + 32;
#pragma unroll
            for (int i = 0; i < 8; ++i) {
                int id = i * 256 + tid;
                int r  = id >> 4, c = (id & 15) * 2;
                areg[i] = *(const float2*)(A + (unsigned)(bm + r) * Kdim + kn + c);
                int k2 = id >> 7, n = id & 127;
                blo[i] = W[(unsigned)(kn + 2 * k2)     * Ndim + bn + n];
                bhi[i] = W[(unsigned)(kn + 2 * k2 + 1) * Ndim + bn + n];
            }
        }

        // A fragments (16x32): elems 0..15 -> K=kb..kb+15 (contiguous per lane half)
        v16bf afrag[2], bfrag[4];
        const int kbA = half * 16;
#pragma unroll
        for (int mt = 0; mt < 2; ++mt) {
            int row = wm * 32 + mt * 16 + l16;
#pragma unroll
            for (int e = 0; e < 8; ++e) {
                afrag[mt][e]     = As[row][kbA + e];
                afrag[mt][8 + e] = As[row][kbA + 8 + e];
            }
        }
        // B fragments (32x16): contiguous 16-K chunk per lane half
        const int kbB = half * 16;
#pragma unroll
        for (int nt = 0; nt < 4; ++nt) {
            int col = wn * 64 + nt * 16 + l16;
#pragma unroll
            for (int e = 0; e < 16; ++e) bfrag[nt][e] = Bs[col][kbB + e];
        }
        // 8 back-to-back WMMAs
#pragma unroll
        for (int nt = 0; nt < 4; ++nt)
#pragma unroll
            for (int mt = 0; mt < 2; ++mt)
                acc[mt][nt] = __builtin_amdgcn_wmma_f32_16x16x32_bf16(
                    false, afrag[mt], false, bfrag[nt], (short)0, acc[mt][nt],
                    false, false);
    }

    // Epilogue (uniform residual branch; 32-bit index math)
    if (resid != nullptr) {
#pragma unroll
        for (int mt = 0; mt < 2; ++mt)
#pragma unroll
            for (int nt = 0; nt < 4; ++nt) {
                int col = bn + wn * 64 + nt * 16 + l16;
                float bv = bias[col];
#pragma unroll
                for (int r = 0; r < 8; ++r) {
                    unsigned row = bm + wm * 32 + mt * 16 + r + half * 8;
                    unsigned idx = row * (unsigned)Ndim + col;
                    C[idx] = acc[mt][nt][r] + bv + resid[idx];
                }
            }
    } else {
#pragma unroll
        for (int mt = 0; mt < 2; ++mt)
#pragma unroll
            for (int nt = 0; nt < 4; ++nt) {
                int col = bn + wn * 64 + nt * 16 + l16;
                float bv = bias[col];
#pragma unroll
                for (int r = 0; r < 8; ++r) {
                    unsigned row = bm + wm * 32 + mt * 16 + r + half * 8;
                    C[row * (unsigned)Ndim + col] = acc[mt][nt][r] + bv;
                }
            }
    }
}

// ===========================================================================
// RoPE + split: qkv[B,N,H*DH*3] (q,k,v interleaved) -> Q,K,V [B,N,D]
// head-contiguous, rotary applied to Q and K. One thread per (b,n,h,dpair).
// ===========================================================================
__global__ __launch_bounds__(256) void rope_split_kernel(
    const float* __restrict__ qkv, const float* __restrict__ kp,
    float* __restrict__ Qo, float* __restrict__ Ko, float* __restrict__ Vo)
{
    unsigned idx = blockIdx.x * 256 + threadIdx.x;
    unsigned dp = idx & 31;
    unsigned h  = (idx >> 5) & 3;
    unsigned n  = (idx >> 7) & 2047;
    unsigned b  = idx >> 18;
    unsigned d0 = dp * 2, d1 = d0 + 1;

    unsigned qo = (b * NSEQ + n) * (3 * DMODEL) + h * (DHEAD * 3);
    float qe = qkv[qo + d0 * 3 + 0], qd = qkv[qo + d1 * 3 + 0];
    float ke = qkv[qo + d0 * 3 + 1], kd = qkv[qo + d1 * 3 + 1];
    float ve = qkv[qo + d0 * 3 + 2], vd = qkv[qo + d1 * 3 + 2];

    unsigned ck = (b * NSEQ + n) * DHEAD;          // kp: [2,B,1,N,DH]
    const unsigned sinoff = BBATCH * NSEQ * DHEAD;
    float ce = kp[ck + d0], co = kp[ck + d1];
    float se = kp[sinoff + ck + d0], so = kp[sinoff + ck + d1];

    // q' = q*cos + rotate_half(q)*sin ; rotate_half: even=-odd, odd=even
    float qre = qe * ce - qd * se;
    float qro = qd * co + qe * so;
    float kre = ke * ce - kd * se;
    float kro = kd * co + ke * so;

    unsigned ob = (b * NSEQ + n) * DMODEL + h * DHEAD;
    Qo[ob + d0] = qre; Qo[ob + d1] = qro;
    Ko[ob + d0] = kre; Ko[ob + d1] = kro;
    Vo[ob + d0] = ve;  Vo[ob + d1] = vd;
}

// ===========================================================================
// Flash attention per (b,h). Q,K,V,O in [B,N,D] head-contiguous layout.
// Grid: (N/128, B*H). 256 threads = 8 waves; each wave owns 16 query rows.
// Key tile 64; S and P*V via WMMA; online softmax via shfl_xor reductions
// across the 16-lane halves (matching the C-fragment row layout).
// ===========================================================================
#define KT 64
__global__ __launch_bounds__(256) void attn_kernel(
    const float* __restrict__ Q, const float* __restrict__ K,
    const float* __restrict__ V, float* __restrict__ O, float scale)
{
    __shared__ __bf16 Kt[KT][DHEAD + 8];        // [key][dh]
    __shared__ __bf16 Vt[DHEAD][KT + 8];        // [dh][key] (transposed)
    __shared__ __bf16 Pt[8][16][KT + 8];        // per-wave P tile

    const int tid  = threadIdx.x;
    const int wave = tid >> 5;
    const int lane = tid & 31;
    const int half = lane >> 4;
    const int l16  = lane & 15;
    const int b = blockIdx.y / NHEAD;
    const int h = blockIdx.y % NHEAD;
    const unsigned base = (unsigned)b * NSEQ * DMODEL + (unsigned)h * DHEAD;
    const int qrow0 = blockIdx.x * 128 + wave * 16;

    // Q fragments: two 16x32 A-frags covering DH=64 (compiler fuses to b128)
    v16bf qf[2];
    {
        const float* qp = Q + base + (unsigned)(qrow0 + l16) * DMODEL;
        const int kb = half * 16;
#pragma unroll
        for (int s = 0; s < 2; ++s)
#pragma unroll
            for (int e = 0; e < 8; ++e) {
                qf[s][e]     = (__bf16)qp[s * 32 + kb + e];
                qf[s][8 + e] = (__bf16)qp[s * 32 + kb + 8 + e];
            }
    }

    v8f acc[4];
#pragma unroll
    for (int nt = 0; nt < 4; ++nt)
        acc[nt] = (v8f){0.f,0.f,0.f,0.f,0.f,0.f,0.f,0.f};
    float mrow[8], lrow[8];
#pragma unroll
    for (int r = 0; r < 8; ++r) { mrow[r] = -1e30f; lrow[r] = 0.f; }

    for (int j0 = 0; j0 < NSEQ; j0 += KT) {
        __syncthreads();
        // Stage K (row-major, packed pairs) and V (transposed), fp32 -> bf16
#pragma unroll
        for (int i = 0; i < 8; ++i) {
            int id = i * 256 + tid;
            int key = id >> 5, d = (id & 31) * 2;
            const float* kp = K + base + (unsigned)(j0 + key) * DMODEL;
            const float* vp = V + base + (unsigned)(j0 + key) * DMODEL;
            float2 kv = *(const float2*)(kp + d);
            float2 vv = *(const float2*)(vp + d);
            bfx2 pk; pk[0] = (__bf16)kv.x; pk[1] = (__bf16)kv.y;
            *(bfx2*)&Kt[key][d] = pk;
            Vt[d][key]     = (__bf16)vv.x;
            Vt[d + 1][key] = (__bf16)vv.y;
        }
        __syncthreads();

        // S = Q @ K^T for 4 key sub-tiles of 16
        float sv[4][8];
        const int kbB = half * 16;
#pragma unroll
        for (int nt = 0; nt < 4; ++nt) {
            v8f s = (v8f){0.f,0.f,0.f,0.f,0.f,0.f,0.f,0.f};
            int key = nt * 16 + l16;
#pragma unroll
            for (int ks = 0; ks < 2; ++ks) {
                v16bf kf;
                int koff = ks * 32 + kbB;
#pragma unroll
                for (int e = 0; e < 16; ++e) kf[e] = Kt[key][koff + e];
                s = __builtin_amdgcn_wmma_f32_16x16x32_bf16(
                    false, qf[ks], false, kf, (short)0, s, false, false);
            }
#pragma unroll
            for (int r = 0; r < 8; ++r) sv[nt][r] = s[r] * scale;
        }

        // Online softmax: row reductions across the 16 lanes of each half
        float alpha[8];
#pragma unroll
        for (int r = 0; r < 8; ++r) {
            float tm = fmaxf(fmaxf(sv[0][r], sv[1][r]),
                             fmaxf(sv[2][r], sv[3][r]));
            tm = fmaxf(tm, __shfl_xor(tm, 1));
            tm = fmaxf(tm, __shfl_xor(tm, 2));
            tm = fmaxf(tm, __shfl_xor(tm, 4));
            tm = fmaxf(tm, __shfl_xor(tm, 8));
            float mnew = fmaxf(mrow[r], tm);
            alpha[r] = __expf(mrow[r] - mnew);
            mrow[r] = mnew;
        }
        float rs[8];
#pragma unroll
        for (int r = 0; r < 8; ++r) rs[r] = 0.f;
#pragma unroll
        for (int nt = 0; nt < 4; ++nt)
#pragma unroll
            for (int r = 0; r < 8; ++r) {
                float p = __expf(sv[nt][r] - mrow[r]);
                rs[r] += p;
                Pt[wave][r + half * 8][nt * 16 + l16] = (__bf16)p;
            }
#pragma unroll
        for (int r = 0; r < 8; ++r) {
            rs[r] += __shfl_xor(rs[r], 1);
            rs[r] += __shfl_xor(rs[r], 2);
            rs[r] += __shfl_xor(rs[r], 4);
            rs[r] += __shfl_xor(rs[r], 8);
            lrow[r] = lrow[r] * alpha[r] + rs[r];
        }
#pragma unroll
        for (int nt = 0; nt < 4; ++nt)
#pragma unroll
            for (int r = 0; r < 8; ++r) acc[nt][r] *= alpha[r];

        __syncthreads();  // make P visible across lanes (uniform trip count)

        // O += P @ V : P as A-frags (16x32 over keys), V as B-frags
        const int kbA = half * 16;
#pragma unroll
        for (int ks = 0; ks < 2; ++ks) {
            v16bf pf;
            int pkb = ks * 32 + kbA;
#pragma unroll
            for (int e = 0; e < 8; ++e) {
                pf[e]     = Pt[wave][l16][pkb + e];
                pf[8 + e] = Pt[wave][l16][pkb + 8 + e];
            }
#pragma unroll
            for (int nt = 0; nt < 4; ++nt) {
                v16bf vf;
                int vkb = ks * 32 + half * 16;
                int vcol = nt * 16 + l16;
#pragma unroll
                for (int e = 0; e < 16; ++e) vf[e] = Vt[vcol][vkb + e];
                acc[nt] = __builtin_amdgcn_wmma_f32_16x16x32_bf16(
                    false, pf, false, vf, (short)0, acc[nt], false, false);
            }
        }
    }

    // Write O = acc / l into head-contiguous [B,N,D] (merged ctx layout)
#pragma unroll
    for (int nt = 0; nt < 4; ++nt)
#pragma unroll
        for (int r = 0; r < 8; ++r) {
            unsigned row = qrow0 + r + half * 8;
            unsigned col = nt * 16 + l16;
            O[base + row * DMODEL + col] = acc[nt][r] / lrow[r];
        }
}

// ===========================================================================
// Row-wise LayerNorm (over 512) + exact GELU, in place. One block per row.
// ===========================================================================
__global__ __launch_bounds__(256) void ln_gelu_kernel(
    float* __restrict__ Hb, const float* __restrict__ g,
    const float* __restrict__ beta)
{
    __shared__ float red[256];
    const int tid = threadIdx.x;
    float* hp = Hb + (size_t)blockIdx.x * 512;
    float x0 = hp[tid], x1 = hp[tid + 256];

    red[tid] = x0 + x1;
    __syncthreads();
#pragma unroll
    for (int st = 128; st > 0; st >>= 1) {
        if (tid < st) red[tid] += red[tid + st];
        __syncthreads();
    }
    float mu = red[0] * (1.0f / 512.0f);
    __syncthreads();

    float d0 = x0 - mu, d1 = x1 - mu;
    red[tid] = d0 * d0 + d1 * d1;
    __syncthreads();
#pragma unroll
    for (int st = 128; st > 0; st >>= 1) {
        if (tid < st) red[tid] += red[tid + st];
        __syncthreads();
    }
    float rstd = rsqrtf(red[0] * (1.0f / 512.0f) + 1e-5f);

    float y0 = d0 * rstd * g[tid] + beta[tid];
    float y1 = d1 * rstd * g[tid + 256] + beta[tid + 256];
    hp[tid]       = 0.5f * y0 * (1.0f + erff(y0 * 0.70710678118654752f));
    hp[tid + 256] = 0.5f * y1 * (1.0f + erff(y1 * 0.70710678118654752f));
}

// ===========================================================================
// cat[m, 0:256] = X[m,:], cat[m, 256:512] = Msg[m,:]
// ===========================================================================
__global__ __launch_bounds__(256) void concat_kernel(
    const float* __restrict__ X, const float* __restrict__ Msg,
    float* __restrict__ Cat)
{
    unsigned idx = blockIdx.x * 256 + threadIdx.x;   // over M*512
    unsigned row = idx >> 9;
    unsigned c = idx & 511;
    Cat[idx] = (c < 256) ? X[row * 256 + c] : Msg[row * 256 + (c - 256)];
}

// ---------------------------------------------------------------------------
static inline void gemm(const float* A, const float* W, const float* bias,
                        const float* resid, float* C, int M, int Kd, int Nd,
                        hipStream_t s)
{
    dim3 grid(Nd / 128, M / 128);
    gemm_bias_kernel<<<grid, 256, 0, s>>>(A, W, bias, resid, C, Kd, Nd);
}

extern "C" void kernel_launch(void* const* d_in, const int* in_sizes, int n_in,
                              void* d_out, int out_size, void* d_ws, size_t ws_size,
                              hipStream_t stream)
{
    (void)in_sizes; (void)n_in; (void)out_size; (void)ws_size;
    const float* kp0     = (const float*)d_in[0];
    const float* kp1     = (const float*)d_in[1];
    const float* desc0   = (const float*)d_in[2];
    const float* desc1   = (const float*)d_in[3];
    const float* sa_Wqkv = (const float*)d_in[4];
    const float* sa_bqkv = (const float*)d_in[5];
    const float* sa_Wo   = (const float*)d_in[6];
    const float* sa_bo   = (const float*)d_in[7];
    const float* sa_W1   = (const float*)d_in[8];
    const float* sa_b1   = (const float*)d_in[9];
    const float* sa_g    = (const float*)d_in[10];
    const float* sa_beta = (const float*)d_in[11];
    const float* sa_W2   = (const float*)d_in[12];
    const float* sa_b2   = (const float*)d_in[13];
    const float* ca_Wqk  = (const float*)d_in[14];
    const float* ca_bqk  = (const float*)d_in[15];
    const float* ca_Wv   = (const float*)d_in[16];
    const float* ca_bv   = (const float*)d_in[17];
    const float* ca_Wo   = (const float*)d_in[18];
    const float* ca_bo   = (const float*)d_in[19];
    const float* ca_W1   = (const float*)d_in[20];
    const float* ca_b1   = (const float*)d_in[21];
    const float* ca_g    = (const float*)d_in[22];
    const float* ca_beta = (const float*)d_in[23];
    const float* ca_W2   = (const float*)d_in[24];
    const float* ca_b2   = (const float*)d_in[25];

    float* out = (float*)d_out;
    float* ws  = (float*)d_ws;
    const int M = MROWS;

    size_t o = 0;
    float* qkv = ws + o; o += (size_t)M * 768;
    float* qb  = ws + o; o += (size_t)M * 256;
    float* kb  = ws + o; o += (size_t)M * 256;
    float* vb  = ws + o; o += (size_t)M * 256;
    float* ctx = ws + o; o += (size_t)M * 256;
    float* msg = ws + o; o += (size_t)M * 256;
    float* cat = ws + o; o += (size_t)M * 512;
    float* hb  = ws + o; o += (size_t)M * 512;
    float* d0p = ws + o; o += (size_t)M * 256;
    float* d1p = ws + o; o += (size_t)M * 256;

    const dim3 agrid(NSEQ / 128, BBATCH * NHEAD);
    const float scale = 0.125f;   // DH^-0.5

    // ---- Self blocks (shared weights) ----
    for (int i = 0; i < 2; ++i) {
        const float* desc = i ? desc1 : desc0;
        const float* kp   = i ? kp1 : kp0;
        float* dres       = i ? d1p : d0p;
        gemm(desc, sa_Wqkv, sa_bqkv, nullptr, qkv, M, 256, 768, stream);
        rope_split_kernel<<<4096, 256, 0, stream>>>(qkv, kp, qb, kb, vb);
        attn_kernel<<<agrid, 256, 0, stream>>>(qb, kb, vb, ctx, scale);
        gemm(ctx, sa_Wo, sa_bo, nullptr, msg, M, 256, 256, stream);
        concat_kernel<<<(M * 512) / 256, 256, 0, stream>>>(desc, msg, cat);
        gemm(cat, sa_W1, sa_b1, nullptr, hb, M, 512, 512, stream);
        ln_gelu_kernel<<<M, 256, 0, stream>>>(hb, sa_g, sa_beta);
        gemm(hb, sa_W2, sa_b2, desc, dres, M, 512, 256, stream);
    }

    // ---- Cross block ----
    float* qk0 = qb; float* qk1 = kb; float* v0 = vb; float* v1 = qkv;
    gemm(d0p, ca_Wqk, ca_bqk, nullptr, qk0, M, 256, 256, stream);
    gemm(d1p, ca_Wqk, ca_bqk, nullptr, qk1, M, 256, 256, stream);
    gemm(d0p, ca_Wv,  ca_bv,  nullptr, v0,  M, 256, 256, stream);
    gemm(d1p, ca_Wv,  ca_bv,  nullptr, v1,  M, 256, 256, stream);

    for (int i = 0; i < 2; ++i) {
        const float* q = i ? qk1 : qk0;
        const float* k = i ? qk0 : qk1;
        const float* v = i ? v0  : v1;
        const float* x = i ? d1p : d0p;
        float* outp = out + (size_t)i * M * 256;
        attn_kernel<<<agrid, 256, 0, stream>>>(q, k, v, ctx, scale);
        gemm(ctx, ca_Wo, ca_bo, nullptr, msg, M, 256, 256, stream);
        concat_kernel<<<(M * 512) / 256, 256, 0, stream>>>(x, msg, cat);
        gemm(cat, ca_W1, ca_b1, nullptr, hb, M, 512, 512, stream);
        ln_gelu_kernel<<<M, 256, 0, stream>>>(hb, ca_g, ca_beta);
        gemm(hb, ca_W2, ca_b2, x, outp, M, 512, 256, stream);
    }
}
